// SelfAttention_4157528342553
// MI455X (gfx1250) — matrix operand order repaired and checked by reading, not compiled, not run on hardware
//
#include <hip/hip_runtime.h>
#include <hip/hip_bf16.h>

typedef __bf16 bf16;
typedef __bf16 v16bf __attribute__((ext_vector_type(16)));
typedef float  v8f   __attribute__((ext_vector_type(8)));
typedef unsigned u32x4 __attribute__((ext_vector_type(4)));
typedef int      i32x4 __attribute__((ext_vector_type(4)));
typedef int      i32x8 __attribute__((ext_vector_type(8)));

#define HIDDEN 2048
#define NHEADS 16
#define HDIM   128
#define BATCH  2
#define SEQ    2048
#define NTOK   (BATCH * SEQ)   // 4096

// The 6-arg tensor_load_to_lds builtin ships with the therock/clang-23
// toolchain, which is also the only one providing this header. ROCm 7.2
// (clang-22) uses the 5-arg form.
#if __has_include(<hip/amd_detail/amd_gfx1250_TDM.h>)
#define TDM_6ARG 1
#endif

// ---------------------------------------------------------------------------
// Helpers
// ---------------------------------------------------------------------------
union Frag {                 // one 16x32 (A) or 32x16 (B) bf16 WMMA fragment
  v16bf  v;
  float4 f4[2];
  unsigned u[8];
};

__device__ __forceinline__ v8f zero8() {
  v8f z;
#pragma unroll
  for (int i = 0; i < 8; ++i) z[i] = 0.0f;
  return z;
}

__device__ __forceinline__ unsigned pack2(float a, float b) {
  union { bf16 h[2]; unsigned u; } t;
  t.h[0] = (bf16)a;
  t.h[1] = (bf16)b;
  return t.u;
}

// Hardware base-2 exponential (v_exp_f32).
__device__ __forceinline__ float fast_exp2(float x) {
  return __builtin_amdgcn_exp2f(x);
}

__device__ __forceinline__ v8f wmma_bf16(v16bf a, v16bf b, v8f c) {
  // D = A(16x32) * B(32x16) + C, fp32 accumulate
  return __builtin_amdgcn_wmma_f32_16x16x32_bf16(
      /*neg_a=*/false, a, /*neg_b=*/false, b,
      /*c_mod=*/(short)0, c, /*reuse_a=*/false, /*reuse_b=*/false);
}

// LDS byte offset of a __shared__ object: generic pointers into the LDS
// aperture carry the wave-relative LDS offset in addr[31:0].
__device__ __forceinline__ unsigned lds_off(const void* p) {
  return (unsigned)(uintptr_t)p;
}

// ---------------------------------------------------------------------------
// Tensor Data Mover: async 2-D bf16 tile load, global -> LDS.
// D# group 1 (loop-invariant part), per CDNA5 ISA sec. 8.4:
//   tile_d0 elements per contiguous row, tile_d1 rows,
//   stride0 elements between rows in global memory,
//   pad_icode: LDS pad interval = 2^(code+1) dwords (set to one full row),
//   pad_acode: LDS pad amount   = code+1 dwords (bank-conflict padding).
// ---------------------------------------------------------------------------
__device__ __forceinline__ i32x8 tdm_group1(unsigned tile_d0, unsigned tile_d1,
                                            unsigned long long stride0,
                                            unsigned pad_icode, unsigned pad_acode) {
  i32x8 g1;
  g1[0] = (int)((1u << 16)                      // data_size = 2 bytes
              | (1u << 20)                      // pad_enable
              | (pad_icode << 22)               // pad_interval
              | (pad_acode << 25));             // pad_amount
  g1[1] = (int)((tile_d0 & 0xFFFFu) << 16);     // tensor_dim0[15:0]
  g1[2] = (int)((tile_d0 >> 16) | ((tile_d1 & 0xFFFFu) << 16)); // td0 hi|td1 lo
  g1[3] = (int)((tile_d1 >> 16) | (tile_d0 << 16));             // td1 hi|tile_dim0
  g1[4] = (int)(tile_d1 & 0xFFFFu);             // tile_dim1 (tile_dim2 = 0)
  g1[5] = (int)(unsigned)(stride0 & 0xFFFFFFFFull);
  g1[6] = (int)(unsigned)((stride0 >> 32) & 0xFFFFull);  // stride0 hi (stride1=0)
  g1[7] = 0;
  return g1;
}

// Issue one TDM descriptor (D# group 0 built per call; groups 2/3 zero = 2-D).
// Tracked with TENSORcnt. Caller must guarantee a *scalar* branch so only one
// wave issues (tensor ops ignore EXEC).
__device__ __forceinline__ void tdm_issue(unsigned lds_byte, const void* gptr,
                                          i32x8 g1) {
  unsigned long long ga = (unsigned long long)(uintptr_t)gptr;
  u32x4 g0;
  g0[0] = 1u;                                   // count=1, user descriptor
  g0[1] = lds_byte;                             // lds_addr
  g0[2] = (unsigned)ga;                         // global_addr lo
  g0[3] = (unsigned)(ga >> 32) | (2u << 30);    // global_addr hi | type=2
  i32x4 z4 = {0, 0, 0, 0};
#ifdef TDM_6ARG
  i32x8 z8 = {0, 0, 0, 0, 0, 0, 0, 0};
  __builtin_amdgcn_tensor_load_to_lds(g0, g1, z4, z4, z8, 0);
#else
  __builtin_amdgcn_tensor_load_to_lds(g0, g1, z4, z4, 0);
#endif
}

// ---------------------------------------------------------------------------
// fp32 -> bf16 conversion (vectorized: float4 in, 2 dwords out)
// ---------------------------------------------------------------------------
__global__ __launch_bounds__(256) void cvt_kernel(const float4* __restrict__ src,
                                                  uint2* __restrict__ dst, int n4) {
  int i = blockIdx.x * 256 + threadIdx.x;
  if (i < n4) {
    float4 f = src[i];
    uint2 o;
    o.x = pack2(f.x, f.y);
    o.y = pack2(f.z, f.w);
    dst[i] = o;
  }
}

// ---------------------------------------------------------------------------
// Tiled WMMA GEMM:  C[m,n] = (sum_k A[m,k] * W[n,k] + bias[n]) * scale
//   A: [M,K] bf16 row-major, W: [N,K] bf16 row-major (Linear weight layout)
//   MODE 0: out bf16, per-head layout [b, h, s, d]       (Q with scale, K)
//   MODE 1: out bf16, per-head transposed [b, h, d, s]   (V^T)
//   MODE 2: out f32, [M, N]                              (final projection)
// Block: 256 threads (8 waves), tile 128x128, K-step 32, TDM double-buffered.
// Wave tile 64x32 -> 4x2 accumulator tiles, 8 WMMAs per K-step.
// LDS tiles row-major-along-K: 32 bf16 (64B) + 16B pad = 80B row pitch.
// ---------------------------------------------------------------------------
template <int MODE>
__global__ __launch_bounds__(256) void gemm_kernel(
    const bf16* __restrict__ A, const bf16* __restrict__ W,
    const float* __restrict__ bias, void* __restrict__ outp,
    int M, int N, int K, float scale) {
  __shared__ __align__(16) bf16 sA[2][128 * 40];
  __shared__ __align__(16) bf16 sB[2][128 * 40];

  const int tid  = threadIdx.x;
  const int lane = tid & 31;
  const int lr   = lane & 15;
  const int hi   = lane >> 4;          // 0: lanes 0-15, 1: lanes 16-31
  const int wave = __builtin_amdgcn_readfirstlane(tid) >> 5;  // scalar wave id
  const int mBase = blockIdx.y * 128;
  const int nBase = blockIdx.x * 128;
  const int wm = (wave >> 2) * 64;     // wave m offset (2 waves in m)
  const int wn = (wave & 3) * 32;      // wave n offset (4 waves in n)

  v8f acc[4][2];
#pragma unroll
  for (int a = 0; a < 4; ++a)
#pragma unroll
    for (int b2 = 0; b2 < 2; ++b2) acc[a][b2] = zero8();

  // Loop-invariant TDM descriptor: 128 rows x 32 elems, row pitch padded 80B.
  const i32x8 gdesc = tdm_group1(32, 128, (unsigned long long)K, 3, 3);

  // Prologue: stage K-step 0 into buffer 0 (scalar branch: wave 0 only).
  if (wave == 0) {
    tdm_issue(lds_off(&sA[0][0]), A + (size_t)mBase * K, gdesc);
    tdm_issue(lds_off(&sB[0][0]), W + (size_t)nBase * K, gdesc);
  }

  const int ksteps = K / 32;
  for (int ks = 0; ks < ksteps; ++ks) {
    const int buf = ks & 1;
    if (wave == 0) {
      if (ks + 1 < ksteps) {
        const int kk2 = (ks + 1) * 32;
        tdm_issue(lds_off(&sA[buf ^ 1][0]), A + (size_t)mBase * K + kk2, gdesc);
        tdm_issue(lds_off(&sB[buf ^ 1][0]), W + (size_t)nBase * K + kk2, gdesc);
        __builtin_amdgcn_s_wait_tensorcnt(2);  // in-order: current buffer done
      } else {
        __builtin_amdgcn_s_wait_tensorcnt(0);
      }
    }
    __syncthreads();

    // A fragment (16x32 bf16): lane<16 -> K dwords {0-3, 8-11}, lane>=16 -> {4-7, 12-15}
    Frag fa[4];
#pragma unroll
    for (int a = 0; a < 4; ++a) {
      const char* p = (const char*)&sA[buf][0] + (wm + 16 * a + lr) * 80 + hi * 16;
      fa[a].f4[0] = *(const float4*)p;
      fa[a].f4[1] = *(const float4*)(p + 32);
    }
    // B fragment (32x16 bf16): lane<16 -> K dwords {0-3, 8-11}, lane>=16 -> {4-7, 12-15}
    Frag fb[2];
#pragma unroll
    for (int b2 = 0; b2 < 2; ++b2) {
      const char* p = (const char*)&sB[buf][0] + (wn + 16 * b2 + lr) * 80 + hi * 16;
      fb[b2].f4[0] = *(const float4*)p;
      fb[b2].f4[1] = *(const float4*)(p + 32);
    }
#pragma unroll
    for (int a = 0; a < 4; ++a)
#pragma unroll
      for (int b2 = 0; b2 < 2; ++b2)
        acc[a][b2] = wmma_bf16(fa[a].v, fb[b2].v, acc[a][b2]);
    __syncthreads();   // protect buffer about to be overwritten by next issue
  }

  // Epilogue. C/D layout: lane<16: n = lr, m = vgpr r; lane>=16: n = lr, m = 8 + r
#pragma unroll
  for (int a = 0; a < 4; ++a) {
#pragma unroll
    for (int b2 = 0; b2 < 2; ++b2) {
      int n = nBase + wn + 16 * b2 + lr;
      float bv = bias[n];
      int mt = mBase + wm + 16 * a + hi * 8;
#pragma unroll
      for (int r = 0; r < 8; ++r) {
        int m = mt + r;
        float v = (acc[a][b2][r] + bv) * scale;
        if (MODE == 2) {
          ((float*)outp)[(size_t)m * N + n] = v;
        } else {
          int b_ = m >> 11;            // token -> batch (S = 2048)
          int s  = m & (SEQ - 1);
          int h  = n >> 7;             // channel -> head (HD = 128)
          int d  = n & (HDIM - 1);
          size_t idx = (MODE == 0)
              ? ((size_t)(b_ * NHEADS + h) * SEQ + s) * HDIM + d
              : ((size_t)(b_ * NHEADS + h) * HDIM + d) * SEQ + s;
          ((bf16*)outp)[idx] = (bf16)v;
        }
      }
    }
  }
}

// ---------------------------------------------------------------------------
// Flash attention. Grid: (SEQ/128, NHEADS, BATCH), block 256 (8 waves).
// Each wave owns a 16-query tile (128 queries/block amortize the K/V DMA).
// Scores computed transposed (S^T = K*Q^T, M=s, N=q) so each lane owns one
// query; O accumulated as O^T (M=d, N=q). Softmax runs in the base-2 domain
// (log2(e) folded into the Q scale). K tile (32x128, 272B pitch) and V^T tile
// (128x32, 80B pitch) staged in LDS by the TDM, double-buffered.
// ---------------------------------------------------------------------------
__global__ __launch_bounds__(256) void attn_kernel(
    const bf16* __restrict__ Q, const bf16* __restrict__ Kmat,
    const bf16* __restrict__ Vt, bf16* __restrict__ Aout) {
  __shared__ __align__(16) bf16 sK[2][32 * 136];
  __shared__ __align__(16) bf16 sV[2][128 * 40];

  const int tid  = threadIdx.x;
  const int lane = tid & 31;
  const int lr   = lane & 15;
  const int hi   = lane >> 4;
  const int wave = __builtin_amdgcn_readfirstlane(tid) >> 5;   // scalar wave id
  const int b = blockIdx.z, h = blockIdx.y;
  const int qblock = blockIdx.x * 128;
  const int q0 = qblock + wave * 16;
  const int ql = q0 + lr;                         // this lane's query row
  const size_t headQK = (size_t)(b * NHEADS + h) * SEQ * HDIM;
  const size_t headV  = (size_t)(b * NHEADS + h) * HDIM * SEQ;
  const float NEGINF = -__builtin_inff();

  // Q^T B-fragments, register-resident for whole kernel (4 x 32-d slices)
  Frag fq[4];
  {
    const char* qrow = (const char*)(Q + headQK + (size_t)ql * HDIM);
#pragma unroll
    for (int k0 = 0; k0 < 4; ++k0) {
      int off = 64 * k0 + hi * 16;
      fq[k0].f4[0] = *(const float4*)(qrow + off);
      fq[k0].f4[1] = *(const float4*)(qrow + off + 32);
    }
  }

  v8f o[8];                      // O^T accumulators: 8 d-chunks of 16
#pragma unroll
  for (int c = 0; c < 8; ++c) o[c] = zero8();
  float mrun = NEGINF, lrun = 0.0f;

  const i32x8 gK = tdm_group1(128, 32, (unsigned long long)HDIM, 5, 3);
  const i32x8 gV = tdm_group1(32, 128, (unsigned long long)SEQ, 3, 3);

  const int nsteps = qblock / 32 + 4;   // causal bound for this block

  // Prologue: stage step 0 into buffer 0 (scalar branch: wave 0 only).
  if (wave == 0) {
    tdm_issue(lds_off(&sK[0][0]), Kmat + headQK, gK);
    tdm_issue(lds_off(&sV[0][0]), Vt + headV, gV);
  }

  for (int st = 0; st < nsteps; ++st) {
    const int sb = st * 32;
    const int buf = st & 1;
    if (wave == 0) {
      if (st + 1 < nsteps) {
        const int sb2 = sb + 32;
        tdm_issue(lds_off(&sK[buf ^ 1][0]), Kmat + headQK + (size_t)sb2 * HDIM, gK);
        tdm_issue(lds_off(&sV[buf ^ 1][0]), Vt + headV + sb2, gV);
        __builtin_amdgcn_s_wait_tensorcnt(2);
      } else {
        __builtin_amdgcn_s_wait_tensorcnt(0);
      }
    }
    __syncthreads();

    // Wave-uniform activity / diagonal tests as scalar branches.
    if (__builtin_amdgcn_readfirstlane((int)(sb <= q0 + 15))) {
      // S^T tiles: d0 covers s in [sb, sb+16), d1 covers [sb+16, sb+32)
      v8f d0 = zero8(), d1 = zero8();
#pragma unroll
      for (int k0 = 0; k0 < 4; ++k0) {
        Frag a0, a1;
        const char* pa = (const char*)&sK[buf][0] + lr * 272 + k0 * 64 + hi * 16;
        a0.f4[0] = *(const float4*)pa;
        a0.f4[1] = *(const float4*)(pa + 32);
        const char* pb_ = (const char*)&sK[buf][0] + (16 + lr) * 272 + k0 * 64 + hi * 16;
        a1.f4[0] = *(const float4*)pb_;
        a1.f4[1] = *(const float4*)(pb_ + 32);
        d0 = wmma_bf16(a0.v, fq[k0].v, d0);
        d1 = wmma_bf16(a1.v, fq[k0].v, d1);
      }

      // Causal mask (diagonal steps only) + online softmax in base-2 domain.
      // Step 0 always has an unmasked element per lane, so m stays finite and
      // exp2(-inf - m) == 0 needs no guards.
      float p0[8], p1[8];
      float tmax = NEGINF;
      if (__builtin_amdgcn_readfirstlane((int)(sb + 31 > q0))) {
#pragma unroll
        for (int r = 0; r < 8; ++r) {
          int s0 = sb + hi * 8 + r;
          p0[r] = (s0 <= ql) ? d0[r] : NEGINF;
          p1[r] = (s0 + 16 <= ql) ? d1[r] : NEGINF;
          tmax = fmaxf(tmax, fmaxf(p0[r], p1[r]));
        }
      } else {
#pragma unroll
        for (int r = 0; r < 8; ++r) {
          p0[r] = d0[r];
          p1[r] = d1[r];
          tmax = fmaxf(tmax, fmaxf(p0[r], p1[r]));
        }
      }
      tmax = fmaxf(tmax, __shfl_xor(tmax, 16, 32));
      float mnew = fmaxf(mrun, tmax);
      float cf = (mrun == mnew) ? 1.0f : fast_exp2(mrun - mnew);
      float rs = 0.0f;
#pragma unroll
      for (int r = 0; r < 8; ++r) {
        p0[r] = fast_exp2(p0[r] - mnew);
        p1[r] = fast_exp2(p1[r] - mnew);
        rs += p0[r] + p1[r];
      }
      rs += __shfl_xor(rs, 16, 32);
      lrun = lrun * cf + rs;
      mrun = mnew;
#pragma unroll
      for (int c = 0; c < 8; ++c) o[c] = o[c] * cf;

      // Build B(P^T) fragment: one xor-16 exchange, then pack bf16 pairs
      float ex0[8], ex1[8];
#pragma unroll
      for (int r = 0; r < 8; ++r) {
        ex0[r] = __shfl_xor(p0[r], 16, 32);
        ex1[r] = __shfl_xor(p1[r], 16, 32);
      }
      Frag pb;
#pragma unroll
      for (int j = 0; j < 4; ++j) {
        float l0 = hi ? ex1[2 * j]     : p0[2 * j];
        float l1 = hi ? ex1[2 * j + 1] : p0[2 * j + 1];
        float h0 = hi ? p1[2 * j]      : ex0[2 * j];
        float h1 = hi ? p1[2 * j + 1]  : ex0[2 * j + 1];
        pb.u[j]     = pack2(l0, l1);
        pb.u[4 + j] = pack2(h0, h1);
      }

      // O^T += V^T(16d x 32s) * P^T(32s x 16q) for 8 d-chunks
#pragma unroll
      for (int c = 0; c < 8; ++c) {
        Frag av;
        const char* p = (const char*)&sV[buf][0] + (16 * c + lr) * 80 + hi * 32;
        av.f4[0] = *(const float4*)p;
        av.f4[1] = *(const float4*)(p + 16);
        o[c] = wmma_bf16(av.v, pb.v, o[c]);
      }
    }
    __syncthreads();
  }

  // Normalize and store bf16 attention output in [B, S, H] token-major layout
  float rl = 1.0f / lrun;
  char* orow = (char*)(Aout + ((size_t)(b * SEQ + ql)) * HIDDEN + (size_t)h * HDIM);
#pragma unroll
  for (int c = 0; c < 8; ++c) {
    union { float4 f4; unsigned u[4]; } pk;
#pragma unroll
    for (int j = 0; j < 4; ++j)
      pk.u[j] = pack2(o[c][2 * j] * rl, o[c][2 * j + 1] * rl);
    *(float4*)(orow + (16 * c + hi * 8) * 2) = pk.f4;
  }
}

// ---------------------------------------------------------------------------
// Launch: cvt x5 -> QKV GEMMs -> flash attention -> output GEMM.
// Workspace: xb 16M + 4xW 32M + Q/K/V^T 48M = 96 MiB (attn output reuses xb).
// ---------------------------------------------------------------------------
extern "C" void kernel_launch(void* const* d_in, const int* in_sizes, int n_in,
                              void* d_out, int out_size, void* d_ws, size_t ws_size,
                              hipStream_t stream) {
  const float* x  = (const float*)d_in[0];
  const float* Wq = (const float*)d_in[1];
  const float* bq = (const float*)d_in[2];
  const float* Wk = (const float*)d_in[3];
  const float* bk = (const float*)d_in[4];
  const float* Wv = (const float*)d_in[5];
  const float* bv = (const float*)d_in[6];
  const float* Wo = (const float*)d_in[7];
  const float* bo = (const float*)d_in[8];

  char* ws = (char*)d_ws;
  size_t off = 0;
  auto carve = [&](size_t bytes) -> char* {
    char* p = ws + off;
    off += (bytes + 255) & ~(size_t)255;
    return p;
  };
  bf16* xb  = (bf16*)carve((size_t)NTOK * HIDDEN * 2);
  bf16* Wqb = (bf16*)carve((size_t)HIDDEN * HIDDEN * 2);
  bf16* Wkb = (bf16*)carve((size_t)HIDDEN * HIDDEN * 2);
  bf16* Wvb = (bf16*)carve((size_t)HIDDEN * HIDDEN * 2);
  bf16* Wob = (bf16*)carve((size_t)HIDDEN * HIDDEN * 2);
  bf16* Qb  = (bf16*)carve((size_t)NTOK * HIDDEN * 2);
  bf16* Kb  = (bf16*)carve((size_t)NTOK * HIDDEN * 2);
  bf16* Vtb = (bf16*)carve((size_t)NTOK * HIDDEN * 2);
  bf16* attnb = xb;   // xb is dead after the QKV GEMMs; reuse for attn output

  const int n4 = NTOK * HIDDEN / 4;
  const int w4 = HIDDEN * HIDDEN / 4;
  cvt_kernel<<<(n4 + 255) / 256, 256, 0, stream>>>((const float4*)x, (uint2*)xb, n4);
  cvt_kernel<<<(w4 + 255) / 256, 256, 0, stream>>>((const float4*)Wq, (uint2*)Wqb, w4);
  cvt_kernel<<<(w4 + 255) / 256, 256, 0, stream>>>((const float4*)Wk, (uint2*)Wkb, w4);
  cvt_kernel<<<(w4 + 255) / 256, 256, 0, stream>>>((const float4*)Wv, (uint2*)Wvb, w4);
  cvt_kernel<<<(w4 + 255) / 256, 256, 0, stream>>>((const float4*)Wo, (uint2*)Wob, w4);

  dim3 ggrid(HIDDEN / 128, NTOK / 128);
  // 1/sqrt(HEAD_DIM) * log2(e): softmax runs in the base-2 domain.
  const float qscale = 0.08838834764831845f * 1.4426950408889634f;
  gemm_kernel<0><<<ggrid, 256, 0, stream>>>(xb, Wqb, bq, Qb, NTOK, HIDDEN, HIDDEN, qscale);
  gemm_kernel<0><<<ggrid, 256, 0, stream>>>(xb, Wkb, bk, Kb, NTOK, HIDDEN, HIDDEN, 1.0f);
  gemm_kernel<1><<<ggrid, 256, 0, stream>>>(xb, Wvb, bv, Vtb, NTOK, HIDDEN, HIDDEN, 1.0f);

  attn_kernel<<<dim3(SEQ / 128, NHEADS, BATCH), 256, 0, stream>>>(Qb, Kb, Vtb, attnb);

  gemm_kernel<2><<<ggrid, 256, 0, stream>>>(attnb, Wob, bo, d_out, NTOK, HIDDEN, HIDDEN, 1.0f);
}
